// LstmLayer_90872918049468
// MI455X (gfx1250) — matrix operand order repaired and checked by reading, not compiled, not run on hardware
//
#include <hip/hip_runtime.h>
#include <cstdint>
#include <cstddef>

// ---------------------------------------------------------------------------
// LSTM layer for MI455X (gfx1250, wave32, WMMA).
// B=32, T=512, I=512, U=1024. Output: h sequence [32,512,1024] f32.
// ---------------------------------------------------------------------------

#define BATCH 32
#define SEQ   512
#define INDIM 512
#define UNITS 1024
#define GATES (4 * UNITS)          // 4096
#define MTOT  (BATCH * SEQ)        // 16384

typedef __attribute__((ext_vector_type(16))) __bf16 bf16x16;
typedef __attribute__((ext_vector_type(8)))  float  f32x8;

union BfVec {
    bf16x16 v;
    uint4   q[2];   // 32 bytes
};

// Workspace layout (bytes)
#define XG_OFF   ((size_t)0)                                   // bf16 x_gates [16384][4096] = 128 MB
#define WHT_OFF  (XG_OFF  + (size_t)MTOT * GATES * 2)          // bf16 Wh^T   [4096][1024]  =   8 MB
#define H_OFF    (WHT_OFF + (size_t)GATES * UNITS * 2)         // bf16 h double buffer 2*[32][1024]
#define SYNC_OFF (H_OFF   + (size_t)2 * BATCH * UNITS * 2)     // 2 x u32 barrier state

// ---- helpers ---------------------------------------------------------------

__device__ __forceinline__ unsigned short f2bf(float f) {
    union { float f; unsigned u; } x; x.f = f;
    unsigned r = x.u + 0x7FFFu + ((x.u >> 16) & 1u);   // round-to-nearest-even
    return (unsigned short)(r >> 16);
}
__device__ __forceinline__ float bf2f(unsigned short s) {
    union { unsigned u; float f; } x; x.u = ((unsigned)s) << 16;
    return x.f;
}
__device__ __forceinline__ float fsigmoid(float x) {
    x = fminf(fmaxf(x, -30.f), 30.f);
    return 1.f / (1.f + __expf(-x));
}
__device__ __forceinline__ float ftanh(float x) {
    x = fminf(fmaxf(x, -15.f), 15.f);
    float e = __expf(2.f * x);
    return (e - 1.f) / (e + 1.f);
}

// Grid-wide split barrier (counter + generation), 32 blocks.
__device__ __forceinline__ void grid_sync(unsigned* cnt, unsigned* gen,
                                          unsigned nblocks, unsigned target) {
    __threadfence();
    __syncthreads();
    if (threadIdx.x == 0) {
        if (atomicAdd(cnt, 1u) == nblocks - 1u) {
            atomicExch(cnt, 0u);
            __threadfence();
            atomicAdd(gen, 1u);
        } else {
            volatile unsigned* vg = gen;
            while (*vg < target) { __builtin_amdgcn_s_sleep(1); }
        }
    }
    __syncthreads();
}

// ---------------------------------------------------------------------------
// Kernel 1: Wh -> bf16 transposed [n][k]; zero h double buffer; reset barrier.
// ---------------------------------------------------------------------------
__global__ __launch_bounds__(256) void prep_kernel(const float* __restrict__ Wh,
                                                   unsigned short* __restrict__ WhT,
                                                   unsigned short* __restrict__ hbuf,
                                                   unsigned* __restrict__ sync) {
    unsigned gid = blockIdx.x * 256u + threadIdx.x;       // [0, 4096*1024)
    unsigned n = gid >> 10;          // 0..4095
    unsigned k = gid & 1023u;        // 0..1023
    WhT[(size_t)n * UNITS + k] = f2bf(Wh[(size_t)k * GATES + n]);
    if (gid < 2u * BATCH * UNITS) hbuf[gid] = 0;          // h0 = 0 (both buffers)
    if (gid < 2u) sync[gid] = 0u;                         // barrier cnt/gen
}

// ---------------------------------------------------------------------------
// Kernel 2: x_gates = data @ Wx + b  (bf16 WMMA, f32 accum, bf16 out).
// Block = 128 threads (4 waves). Block tile: 16(M) x 64(N). K loop of 32.
// grid = (16384/16, 4096/64) = (1024, 64)
// ---------------------------------------------------------------------------
__global__ __launch_bounds__(128) void xproj_kernel(const float* __restrict__ data,
                                                    const float* __restrict__ Wx,
                                                    const float* __restrict__ bias,
                                                    unsigned short* __restrict__ xg) {
    __shared__ __align__(16) unsigned short As[16 * 32];   // [row][k] bf16
    __shared__ __align__(16) unsigned short Bs[64 * 32];   // [n][k]   bf16 (transposed)

    const int tid  = threadIdx.x;
    const int wave = tid >> 5;
    const int lane = tid & 31;
    const int col  = lane & 15;
    const int hi   = lane >> 4;
    const int m0   = blockIdx.x * 16;
    const int n0   = blockIdx.y * 64;

    f32x8 acc = {};

    for (int k0 = 0; k0 < INDIM; k0 += 32) {
        __syncthreads();
        // Stage A tile 16x32 (512 elems, 4/thread) as bf16, row-major.
#pragma unroll
        for (int j = 0; j < 4; ++j) {
            int e = tid * 4 + j;
            int r = e >> 5, k = e & 31;
            As[r * 32 + k] = f2bf(data[(size_t)(m0 + r) * INDIM + k0 + k]);
        }
        // Stage B tile 32x64 (2048 elems, 16/thread) transposed to [n][k].
#pragma unroll
        for (int j = 0; j < 16; ++j) {
            int e = tid * 16 + j;
            int k = e >> 6, n = e & 63;
            Bs[n * 32 + k] = f2bf(Wx[(size_t)(k0 + k) * GATES + n0 + n]);
        }
        __syncthreads();

        // Per-lane WMMA fragments (ISA 16-bit A 16x32 / B 32x16 layouts).
        BfVec a, b;
        const uint4* As4 = (const uint4*)As;
        const uint4* Bs4 = (const uint4*)Bs;
        {   // A: lane holds row=col; k-halves select by hi
            int row = col;
            a.q[0] = As4[row * 4 + hi];          // K = hi*8 .. +7
            a.q[1] = As4[row * 4 + hi + 2];      // K = 16+hi*8 .. +7
        }
        {   // B: lane holds col n; K = hi*8..+7 and 16+hi*8..+7
            int n = wave * 16 + col;
            b.q[0] = Bs4[n * 4 + hi];
            b.q[1] = Bs4[n * 4 + hi + 2];
        }
        acc = __builtin_amdgcn_wmma_f32_16x16x32_bf16(false, a.v, false, b.v,
                                                      (short)0, acc, false, false);
    }

    // C layout: lane col = lane&15, rows = v + 8*hi. Add bias, store bf16 NT.
    const int n = n0 + wave * 16 + col;
    const float bv = bias[n];
#pragma unroll
    for (int v = 0; v < 8; ++v) {
        int m = m0 + v + 8 * hi;
        __builtin_nontemporal_store(f2bf(acc[v] + bv), &xg[(size_t)m * GATES + n]);
    }
}

// ---------------------------------------------------------------------------
// Kernel 3: persistent LSTM recurrence. 32 blocks x 128 threads (128 waves).
// Wave owns one 16x16 unit tile (i,f,g,o accumulated together); c in regs.
// Double-buffered h (bf16); one grid barrier per timestep.
// ---------------------------------------------------------------------------
__global__ __launch_bounds__(128) void lstm_rec_kernel(const unsigned short* __restrict__ xg,
                                                       const unsigned short* __restrict__ WhT,
                                                       unsigned short* __restrict__ hbuf,
                                                       unsigned* __restrict__ sync,
                                                       float* __restrict__ out) {
    const int tid  = threadIdx.x;
    const int wave = tid >> 5;
    const int lane = tid & 31;
    const int col  = lane & 15;
    const int hi   = lane >> 4;

    const int rowTile  = blockIdx.x & 1;                   // 0..1  (batch halves)
    const int unitTile = (blockIdx.x >> 1) * 4 + wave;     // 0..63
    const int u        = unitTile * 16 + col;              // unit column
    const int aRow     = rowTile * 16 + col;               // h row this lane loads (A frag)

    unsigned* cnt = sync;
    unsigned* gen = sync + 1;

    float c8[8];
#pragma unroll
    for (int v = 0; v < 8; ++v) c8[v] = 1.0f;              // c0 = ones (per reference)

    for (int t = 0; t < SEQ; ++t) {
        const unsigned short* hin  = hbuf + (size_t)(t & 1) * (BATCH * UNITS);
        unsigned short*       hout = hbuf + (size_t)((t + 1) & 1) * (BATCH * UNITS);

        f32x8 acc[4];
        acc[0] = (f32x8){}; acc[1] = (f32x8){}; acc[2] = (f32x8){}; acc[3] = (f32x8){};

        for (int kk = 0; kk < UNITS; kk += 32) {
            BfVec a;
            const uint4* hp = (const uint4*)(hin + (size_t)aRow * UNITS + kk);
            a.q[0] = hp[hi];
            a.q[1] = hp[hi + 2];
#pragma unroll
            for (int q = 0; q < 4; ++q) {
                const int n = q * UNITS + u;
                const uint4* wp = (const uint4*)(WhT + (size_t)n * UNITS + kk);
                BfVec b;
                b.q[0] = wp[hi];
                b.q[1] = wp[hi + 2];
                if (kk + 32 < UNITS) __builtin_prefetch(wp + 4, 0, 3);  // next K frag
                acc[q] = __builtin_amdgcn_wmma_f32_16x16x32_bf16(false, a.v, false, b.v,
                                                                 (short)0, acc[q], false, false);
            }
        }

        // Pointwise LSTM cell; C layout rows = v + 8*hi, col = u.
#pragma unroll
        for (int v = 0; v < 8; ++v) {
            const int batch = rowTile * 16 + v + 8 * hi;
            const unsigned short* xr = xg + ((size_t)batch * SEQ + t) * GATES;
            float gi = acc[0][v] + bf2f(xr[0 * UNITS + u]);
            float gf = acc[1][v] + bf2f(xr[1 * UNITS + u]);
            float gg = acc[2][v] + bf2f(xr[2 * UNITS + u]);
            float go = acc[3][v] + bf2f(xr[3 * UNITS + u]);
            float i = fsigmoid(gi);
            float f = fsigmoid(gf);
            float g = ftanh(gg);
            float o = fsigmoid(go);
            float cn = f * c8[v] + i * g;
            c8[v] = cn;
            float hn = o * ftanh(cn);
            __builtin_nontemporal_store(hn, &out[((size_t)batch * SEQ + t) * UNITS + u]);
            hout[(size_t)batch * UNITS + u] = f2bf(hn);
        }

        grid_sync(cnt, gen, gridDim.x, (unsigned)(t + 1));
    }
}

// ---------------------------------------------------------------------------
// Launch
// ---------------------------------------------------------------------------
extern "C" void kernel_launch(void* const* d_in, const int* in_sizes, int n_in,
                              void* d_out, int out_size, void* d_ws, size_t ws_size,
                              hipStream_t stream) {
    (void)in_sizes; (void)n_in; (void)out_size; (void)ws_size;
    const float* data = (const float*)d_in[0];   // [32,512,512]
    const float* Wx   = (const float*)d_in[1];   // [512,4096]
    const float* Wh   = (const float*)d_in[2];   // [1024,4096]
    const float* bias = (const float*)d_in[3];   // [4096]

    char* ws = (char*)d_ws;
    unsigned short* xg   = (unsigned short*)(ws + XG_OFF);
    unsigned short* WhT  = (unsigned short*)(ws + WHT_OFF);
    unsigned short* hbuf = (unsigned short*)(ws + H_OFF);
    unsigned*       sync = (unsigned*)(ws + SYNC_OFF);

    // 1) Wh transpose/convert + state/barrier init
    prep_kernel<<<(GATES * UNITS) / 256, 256, 0, stream>>>(Wh, WhT, hbuf, sync);

    // 2) x-projection GEMM (bf16 WMMA)
    dim3 g1(MTOT / 16, GATES / 64);
    xproj_kernel<<<g1, 128, 0, stream>>>(data, Wx, bias, xg);

    // 3) persistent recurrence
    lstm_rec_kernel<<<32, 128, 0, stream>>>(xg, WhT, hbuf, sync, (float*)d_out);
}
